// GRULearnerLayer_74801150427501
// MI455X (gfx1250) — matrix operand order repaired and checked by reading, not compiled, not run on hardware
//
#include <hip/hip_runtime.h>

// ---------------------------------------------------------------------------
// GRU layer for gfx1250 (MI455X): bf16 WMMA GEMMs with f32 accumulate.
// N=2048 batch, T=64 steps, D=512 hidden.
//
// Per-step fused kernel: each workgroup owns 32 rows x all 512 cols so the
// r*h -> (r*h)@Whh.T dependency is resolved with one LDS barrier, and each
// weight (B) fragment read from L2 feeds TWO row-tiles (halves L2 traffic,
// the dominant bottleneck: weights stream from the 192MB L2 every step).
//
// Three symmetric dual-GEMM passes keep accumulator pressure at 64 VGPRs:
//   pass Z: x@Wzx.T + h@Wzh.T  -> z = sigmoid(+bz)  -> LDS (f32)
//   pass R: x@Wrx.T + h@Wrh.T  -> rh = sigmoid(+br)*h -> LDS (bf16)
//   barrier
//   pass H: x@Whx.T + rh@Whh.T -> h' = z*h + (1-z)*tanh(+bh)
// ---------------------------------------------------------------------------

typedef __attribute__((ext_vector_type(16))) __bf16 v16bf;
typedef __attribute__((ext_vector_type(8)))  __bf16 v8bf;
typedef __attribute__((ext_vector_type(8)))  float  v8f;
typedef __attribute__((ext_vector_type(4)))  float  v4f;

#define NB    2048
#define TT    64
#define DD    512
#define BM    32           // rows per workgroup
#define PITCH 520          // bf16 elements per LDS row (16B-aligned, bank-rotated)
#define ZPITCH 516         // f32 elements per LDS row for z tile

static __device__ __forceinline__ float sigmoid_f(float x) {
    return 1.0f / (1.0f + __expf(-x));     // exp=+inf -> 0; no NaN path
}
static __device__ __forceinline__ float tanh_f(float x) {
    float e = __expf(2.0f * x);            // e=inf -> 1, e=0 -> -1; no NaN path
    return 1.0f - 2.0f / (e + 1.0f);
}

// ----------------------------- setup kernels -------------------------------

__global__ void cvt_w_bf16(const float* __restrict__ src, __bf16* __restrict__ dst) {
    int i = (blockIdx.x * 256 + threadIdx.x) * 4;
    v4f v = *(const v4f*)(src + i);
    dst[i + 0] = (__bf16)v[0];
    dst[i + 1] = (__bf16)v[1];
    dst[i + 2] = (__bf16)v[2];
    dst[i + 3] = (__bf16)v[3];
}

__global__ void zero_f32(float* __restrict__ p) {
    int i = blockIdx.x * 256 + threadIdx.x;
    p[i] = 0.0f;
}

// ------------------------------ step kernel --------------------------------

__global__ __launch_bounds__(256)
void gru_step(const float* __restrict__ x,        // [N,T,D] f32
              const __bf16* __restrict__ Wzx, const __bf16* __restrict__ Wzh,
              const __bf16* __restrict__ Wrx, const __bf16* __restrict__ Wrh,
              const __bf16* __restrict__ Whx, const __bf16* __restrict__ Whh,
              const float* __restrict__ bz, const float* __restrict__ br,
              const float* __restrict__ bh,
              float* __restrict__ h,               // [N,D] f32 state
              float* __restrict__ out,             // [N,T,D] f32
              int t)
{
    __shared__ __bf16 xa[BM * PITCH];   // x_t tile (bf16)          33,280 B
    __shared__ __bf16 ha[BM * PITCH];   // h tile (bf16)            33,280 B
    __shared__ __bf16 rh[BM * PITCH];   // r*h tile (bf16)          33,280 B
    __shared__ float  za[BM * ZPITCH];  // z tile (f32)             66,048 B

    const int tid  = threadIdx.x;
    const int lane = tid & 31;
    const int w    = tid >> 5;            // wave id 0..7
    const int row0 = blockIdx.x * BM;     // first batch row of this WG

    // ---- stage x_t and h tiles: f32 global -> bf16 LDS (coalesced 128-bit) ----
    {
        const int r  = tid >> 3;          // 0..31
        const int c0 = (tid & 7) * 64;    // 0..448
        const float* xg = x + ((size_t)(row0 + r) * TT + t) * DD + c0;
        const float* hg = h + (size_t)(row0 + r) * DD + c0;
        #pragma unroll
        for (int j = 0; j < 64; j += 8) {
            v4f x0 = *(const v4f*)(xg + j);
            v4f x1 = *(const v4f*)(xg + j + 4);
            v4f h0 = *(const v4f*)(hg + j);
            v4f h1 = *(const v4f*)(hg + j + 4);
            v8bf xv, hv;
            #pragma unroll
            for (int e = 0; e < 4; ++e) {
                xv[e]     = (__bf16)x0[e];
                xv[e + 4] = (__bf16)x1[e];
                hv[e]     = (__bf16)h0[e];
                hv[e + 4] = (__bf16)h1[e];
            }
            *(v8bf*)(&xa[r * PITCH + c0 + j]) = xv;
            *(v8bf*)(&ha[r * PITCH + c0 + j]) = hv;
        }
        // prefetch next timestep's x rows into cache (global_prefetch_b8)
        if (t + 1 < TT) {
            const float* xn = x + ((size_t)(row0 + r) * TT + (t + 1)) * DD + c0;
            __builtin_prefetch(xn, 0, 1);
            __builtin_prefetch(xn + 32, 0, 1);
        }
    }
    __syncthreads();

    // A fragment (16x32 bf16): lane row m = L%16, khi=(L>>4)*16,
    // elements 0..15 = A[m][k+khi .. +15] (matches B's contiguous K order)
    union FragU { v16bf v; v8bf half[2]; };
    auto loadA = [&](const __bf16* base, int rt, int kc) -> v16bf {
        const int rA  = rt * 16 + (lane & 15);
        const int khi = (lane >> 4) << 4;
        const __bf16* p = base + rA * PITCH + kc * 32 + khi;
        FragU f;
        f.half[0] = *(const v8bf*)(p);
        f.half[1] = *(const v8bf*)(p + 8);
        return f.v;
    };
    // B fragment (32x16 bf16): B[k][n] = W[n][k]; lane col n = L%16 holds
    // K = (L>>4)*16 .. +15  -> one contiguous 32-byte load of weight row n.
    auto loadB = [&](const __bf16* W, int nn, int kc) -> v16bf {
        const int n = nn + (lane & 15);
        const int k = kc * 32 + ((lane >> 4) << 4);
        return *(const v16bf*)(W + (size_t)n * DD + k);
    };

    const int n0  = w * 64;               // this wave's first output column
    const int mhi = (lane >> 4) << 3;     // C/D layout: M = i + 8*(lane/16)
    const int nl  = lane & 15;            //             N = lane%16

    // dual-matrix GEMM pass: acc[rt][nt] += A0@W0.T + A1@W1.T
    // (B fragments loaded once per (nt,kc), reused for both row-tiles)
    auto gemm_pass = [&](const __bf16* A0, const __bf16* A1,
                         const __bf16* W0, const __bf16* W1,
                         v8f (&acc)[2][4]) {
        for (int kc = 0; kc < 16; ++kc) {
            v16bf a0[2], a1[2];
            a0[0] = loadA(A0, 0, kc); a0[1] = loadA(A0, 1, kc);
            a1[0] = loadA(A1, 0, kc); a1[1] = loadA(A1, 1, kc);
            #pragma unroll
            for (int nt = 0; nt < 4; ++nt) {
                const int nn = n0 + nt * 16;
                v16bf b0 = loadB(W0, nn, kc);
                v16bf b1 = loadB(W1, nn, kc);
                #pragma unroll
                for (int rt = 0; rt < 2; ++rt) {
                    acc[rt][nt] = __builtin_amdgcn_wmma_f32_16x16x32_bf16(
                        false, a0[rt], false, b0, (short)0, acc[rt][nt], false, false);
                    acc[rt][nt] = __builtin_amdgcn_wmma_f32_16x16x32_bf16(
                        false, a1[rt], false, b1, (short)0, acc[rt][nt], false, false);
                }
            }
        }
    };

    // ---------------- pass Z: z = sigmoid(x@Wzx.T + h@Wzh.T + bz) ----------
    {
        v8f acc[2][4] = {};
        gemm_pass(xa, ha, Wzx, Wzh, acc);
        #pragma unroll
        for (int nt = 0; nt < 4; ++nt) {
            const int col = n0 + nt * 16 + nl;
            const float bzv = bz[col];
            #pragma unroll
            for (int rt = 0; rt < 2; ++rt)
                #pragma unroll
                for (int i = 0; i < 8; ++i) {
                    const int m = rt * 16 + mhi + i;
                    za[m * ZPITCH + col] = sigmoid_f(acc[rt][nt][i] + bzv);
                }
        }
    }

    // ---------------- pass R: rh = sigmoid(x@Wrx.T + h@Wrh.T + br) * h -----
    {
        v8f acc[2][4] = {};
        gemm_pass(xa, ha, Wrx, Wrh, acc);
        #pragma unroll
        for (int nt = 0; nt < 4; ++nt) {
            const int col = n0 + nt * 16 + nl;
            const float brv = br[col];
            #pragma unroll
            for (int rt = 0; rt < 2; ++rt)
                #pragma unroll
                for (int i = 0; i < 8; ++i) {
                    const int m = rt * 16 + mhi + i;
                    const float rr = sigmoid_f(acc[rt][nt][i] + brv);
                    const float hv = h[(size_t)(row0 + m) * DD + col];
                    rh[m * PITCH + col] = (__bf16)(rr * hv);
                }
        }
    }
    __syncthreads();   // rh complete across all 8 waves before pass H

    // ---------------- pass H: h' = z*h + (1-z)*tanh(x@Whx.T + rh@Whh.T + bh)
    {
        v8f acc[2][4] = {};
        gemm_pass(xa, rh, Whx, Whh, acc);
        #pragma unroll
        for (int nt = 0; nt < 4; ++nt) {
            const int col = n0 + nt * 16 + nl;
            const float bhv = bh[col];
            #pragma unroll
            for (int rt = 0; rt < 2; ++rt)
                #pragma unroll
                for (int i = 0; i < 8; ++i) {
                    const int m  = rt * 16 + mhi + i;
                    const size_t rg = (size_t)(row0 + m);
                    const float ht = tanh_f(acc[rt][nt][i] + bhv);
                    const float hv = h[rg * DD + col];
                    const float z  = za[m * ZPITCH + col];
                    const float hn = z * hv + (1.0f - z) * ht;
                    h[rg * DD + col] = hn;
                    out[(rg * TT + t) * DD + col] = hn;
                }
        }
    }
}

// ------------------------------- launcher ----------------------------------

extern "C" void kernel_launch(void* const* d_in, const int* in_sizes, int n_in,
                              void* d_out, int out_size, void* d_ws, size_t ws_size,
                              hipStream_t stream) {
    const float* x   = (const float*)d_in[0];
    const float* Wf[6] = { (const float*)d_in[1], (const float*)d_in[2],
                           (const float*)d_in[3], (const float*)d_in[4],
                           (const float*)d_in[5], (const float*)d_in[6] };
    const float* bz = (const float*)d_in[7];
    const float* br = (const float*)d_in[8];
    const float* bh = (const float*)d_in[9];
    float* out = (float*)d_out;

    // workspace: [6 * 512*512 bf16 weights][2048*512 f32 h-state]  (~7.3 MB)
    const size_t WN = (size_t)DD * DD;                 // 262144 elements / matrix
    __bf16* wb = (__bf16*)d_ws;
    float*  h  = (float*)((char*)d_ws + 6 * WN * sizeof(__bf16));

    // one-time (per launch) conversions/initialization — deterministic
    for (int m = 0; m < 6; ++m)
        cvt_w_bf16<<<(int)(WN / (256 * 4)), 256, 0, stream>>>(Wf[m], wb + m * WN);
    zero_f32<<<(NB * DD) / 256, 256, 0, stream>>>(h);

    // sequential recurrence: 64 dependent step launches on the same stream
    for (int t = 0; t < TT; ++t) {
        gru_step<<<NB / BM, 256, 0, stream>>>(
            x, wb + 0 * WN, wb + 1 * WN, wb + 2 * WN, wb + 3 * WN,
            wb + 4 * WN, wb + 5 * WN, bz, br, bh, h, out, t);
    }
}
